// QuantumSpectralConv_55972013801698
// MI455X (gfx1250) — matrix operand order repaired and checked by reading, not compiled, not run on hardware
//
#include <hip/hip_runtime.h>
#include <hip/hip_bf16.h>

// Problem constants (from reference)
#define B_    4
#define HW    128
#define C_    64
#define MODES 16
#define NC    1024   // MODES * C (contraction n,c)
#define MO    1024   // MODES * O (output m,o)
#define TWO_PI_OVER_128 0.04908738521234052f

typedef __attribute__((ext_vector_type(16))) __bf16 v16bf;
typedef __attribute__((ext_vector_type(8)))  float  v8f;

__device__ __forceinline__ unsigned short f2bf(float f) {
  unsigned u = __float_as_uint(f);
  u += 0x7FFFu + ((u >> 16) & 1u);          // round-to-nearest-even
  return (unsigned short)(u >> 16);
}

union FragU { uint4 q[2]; v16bf v; };

__device__ __forceinline__ v16bf ld_frag2(const unsigned short* p0, const unsigned short* p1) {
  FragU u;
  u.q[0] = *reinterpret_cast<const uint4*>(p0);   // 16B-aligned by construction
  u.q[1] = *reinterpret_cast<const uint4*>(p1);
  return u.v;
}

// ---------------------------------------------------------------------------
// Stage 0: Schmidt reduction fused into the single mandatory weight pass
// (33.5 MB read -> this kernel IS the runtime at 23.3 TB/s).
// W_eff[m,n,c,o] = sum_s W[m,n,c,o,s] / sqrt(4); store bf16 as B^T: [mo][nc].
// Grid-stride x4 with global_prefetch_b8 read-ahead; s innermost -> float4 loads.
// ---------------------------------------------------------------------------
#define SUMW_STRIDE (1024 * 256)   // blocks * threads

__global__ __launch_bounds__(256)
void k_sumw(const float* __restrict__ wr, const float* __restrict__ wi,
            unsigned short* __restrict__ Wr16, unsigned short* __restrict__ Wi16) {
  int t = blockIdx.x * 256 + threadIdx.x;
  const float4* wr4 = reinterpret_cast<const float4*>(wr);
  const float4* wi4 = reinterpret_cast<const float4*>(wi);
#pragma unroll
  for (int it = 0; it < 4; ++it, t += SUMW_STRIDE) {
    if (it < 3) {                                   // read-ahead next stride
      __builtin_prefetch(wr4 + t + SUMW_STRIDE, 0, 1);
      __builtin_prefetch(wi4 + t + SUMW_STRIDE, 0, 1);
    }
    const float4 r  = wr4[t];
    const float4 im = wi4[t];
    const float sr = (r.x + r.y + r.z + r.w)     * 0.5f;
    const float si = (im.x + im.y + im.z + im.w) * 0.5f;
    const int o = t & 63, c = (t >> 6) & 63, n = (t >> 12) & 15, m = t >> 16;
    const int oidx = (m * 64 + o) * NC + n * 64 + c;     // [mo][nc]
    Wr16[oidx] = f2bf(sr);
    Wi16[oidx] = f2bf(si);
  }
}

// ---------------------------------------------------------------------------
// Stage 1: truncated DFT along W: Y[b,h,k2,c] = sum_w x[b,h,w,c] e^{-2pi i k2 w/128}
// ---------------------------------------------------------------------------
__global__ __launch_bounds__(256)
void k_dftw(const float* __restrict__ x, float* __restrict__ Yre, float* __restrict__ Yim) {
  __shared__ float ct[MODES * HW];
  __shared__ float st[MODES * HW];
  for (int i = threadIdx.x; i < MODES * HW; i += 256) {
    const int k = i >> 7, w = i & 127;
    float s, c;
    __sincosf((float)((k * w) & 127) * -TWO_PI_OVER_128, &s, &c);
    ct[i] = c; st[i] = s;
  }
  __syncthreads();
  const int t = blockIdx.x * 256 + threadIdx.x;        // [b][h][k2][c]
  const int c = t & 63, k2 = (t >> 6) & 15, h = (t >> 10) & 127, b = t >> 17;
  const float* xp = x + ((b * HW + h) * HW) * 64 + c;
  const float* cw = ct + k2 * HW;
  const float* sw = st + k2 * HW;
  float ar = 0.f, ai = 0.f;
#pragma unroll 4
  for (int w = 0; w < HW; ++w) {
    const float xv = xp[w * 64];
    ar = fmaf(xv, cw[w], ar);
    ai = fmaf(xv, sw[w], ai);
  }
  Yre[t] = ar; Yim[t] = ai;
}

// ---------------------------------------------------------------------------
// Stage 2: truncated DFT along H: X[b,k1,k2,c] = sum_h e^{-2pi i k1 h/128} Y[b,h,k2,c]
// Store bf16 in WMMA-A row-major layout: rows bk = b*16+k1, cols nc = k2*64+c.
// ---------------------------------------------------------------------------
__global__ __launch_bounds__(256)
void k_dfth(const float* __restrict__ Yre, const float* __restrict__ Yim,
            unsigned short* __restrict__ Xr16, unsigned short* __restrict__ Xi16) {
  __shared__ float ct[MODES * HW];
  __shared__ float st[MODES * HW];
  for (int i = threadIdx.x; i < MODES * HW; i += 256) {
    const int k = i >> 7, h = i & 127;
    float s, c;
    __sincosf((float)((k * h) & 127) * -TWO_PI_OVER_128, &s, &c);
    ct[i] = c; st[i] = s;
  }
  __syncthreads();
  const int t = blockIdx.x * 256 + threadIdx.x;        // [b][k1][k2][c]
  const int c = t & 63, k2 = (t >> 6) & 15, k1 = (t >> 10) & 15, b = t >> 14;
  const int ybase = b * (HW * MODES * 64) + k2 * 64 + c;
  const float* ch = ct + k1 * HW;
  const float* sh = st + k1 * HW;
  float ar = 0.f, ai = 0.f;
#pragma unroll 4
  for (int h = 0; h < HW; ++h) {
    const float yr = Yre[ybase + h * 1024];
    const float yi = Yim[ybase + h * 1024];
    const float cc = ch[h], ss = sh[h];
    ar = fmaf(yr, cc, fmaf(-yi, ss, ar));
    ai = fmaf(yr, ss, fmaf( yi, cc, ai));
  }
  const int xi = (b * MODES + k1) * NC + k2 * 64 + c;
  Xr16[xi] = f2bf(ar);
  Xi16[xi] = f2bf(ai);
}

// ---------------------------------------------------------------------------
// Stage 3: complex spectral GEMM on WMMA.
// Z[bk, mo] = sum_nc X[bk,nc] * W_eff[nc,mo];  A: [64 x 1024], B^T rows: [mo][nc].
// One wave per 16x16 output tile (4 x 64 tiles = 256 waves); K-loop of 32 steps,
// 4 x v_wmma_f32_16x16x32_bf16 per step into 4 INDEPENDENT accumulators
// (RR, II, RI, IR) -> hot loop is pure load + wmma, no VALU, no hazard NOPs.
// Combine Re = RR - II, Im = RI + IR once at the end.
// ---------------------------------------------------------------------------
__global__ __launch_bounds__(256)
void k_gemm(const unsigned short* __restrict__ Xr, const unsigned short* __restrict__ Xi,
            const unsigned short* __restrict__ Wr, const unsigned short* __restrict__ Wi,
            float* __restrict__ Zr, float* __restrict__ Zi) {
  const int lane = threadIdx.x & 31;
  const int wave = threadIdx.x >> 5;
  const int gw   = blockIdx.x * 8 + wave;    // 0..255
  const int rt   = gw >> 6;                  // row tile (bk)  0..3
  const int ctl  = gw & 63;                  // col tile (mo)  0..63
  const int hf   = lane >> 4;
  const int l15  = lane & 15;

  // A fragment (16x32 bf16): lane row = l15; K chunks at hf*8 and hf*8+16
  const unsigned short* arow_r = Xr + (rt * 16 + l15) * NC;
  const unsigned short* arow_i = Xi + (rt * 16 + l15) * NC;
  // B fragment (32x16 bf16): lane col = l15; K chunks at hf*8 and hf*8+16
  const unsigned short* bcol_r = Wr + (ctl * 16 + l15) * NC;
  const unsigned short* bcol_i = Wi + (ctl * 16 + l15) * NC;

  v8f aRR = {}, aII = {}, aRI = {}, aIR = {};

  for (int kk = 0; kk < NC / 32; ++kk) {
    const int ka = kk * 32 + hf * 8;
    const int kb = kk * 32 + hf * 8;
    v16bf ar = ld_frag2(arow_r + ka, arow_r + ka + 16);
    v16bf ai = ld_frag2(arow_i + ka, arow_i + ka + 16);
    v16bf br = ld_frag2(bcol_r + kb, bcol_r + kb + 16);
    v16bf bi = ld_frag2(bcol_i + kb, bcol_i + kb + 16);
    aRR = __builtin_amdgcn_wmma_f32_16x16x32_bf16(false, ar, false, br, (short)0, aRR, false, false);
    aII = __builtin_amdgcn_wmma_f32_16x16x32_bf16(false, ai, false, bi, (short)0, aII, false, false);
    aRI = __builtin_amdgcn_wmma_f32_16x16x32_bf16(false, ar, false, bi, (short)0, aRI, false, false);
    aIR = __builtin_amdgcn_wmma_f32_16x16x32_bf16(false, ai, false, br, (short)0, aIR, false, false);
  }

  // C/D layout: VGPR v -> row v + 8*hf, col = l15 (lanes 16-31 carry rows 8..15)
  const int row0 = rt * 16 + hf * 8;
  const int col  = ctl * 16 + l15;
#pragma unroll
  for (int v = 0; v < 8; ++v) {
    Zr[(row0 + v) * MO + col] = aRR[v] - aII[v];
    Zi[(row0 + v) * MO + col] = aRI[v] + aIR[v];
  }
}

// ---------------------------------------------------------------------------
// Stage 4: inverse DFT over m -> w: T[b,k1,w,o] = sum_m Z[b,k1,m,o] e^{+2pi i m w/128}
// ---------------------------------------------------------------------------
__global__ __launch_bounds__(256)
void k_idftm(const float* __restrict__ Zre, const float* __restrict__ Zim,
             float* __restrict__ Tre, float* __restrict__ Tim) {
  const int t = blockIdx.x * 256 + threadIdx.x;        // [b][k1][w][o]
  const int o = t & 63, w = (t >> 6) & 127, k1 = (t >> 13) & 15, b = t >> 17;
  const int zbase = (b * MODES + k1) * MO + o;
  float ar = 0.f, ai = 0.f;
#pragma unroll
  for (int m = 0; m < MODES; ++m) {
    float s, cc;
    __sincosf((float)((m * w) & 127) * TWO_PI_OVER_128, &s, &cc);
    const float zr = Zre[zbase + m * 64];
    const float zi = Zim[zbase + m * 64];
    ar = fmaf(zr, cc, fmaf(-zi, s,  ar));
    ai = fmaf(zr, s,  fmaf( zi, cc, ai));
  }
  Tre[t] = ar; Tim[t] = ai;
}

// ---------------------------------------------------------------------------
// Stage 5: inverse DFT over k1 -> h, take real part, apply 1/(128*128).
// out[b,h,w,o] = Re( sum_k1 T[b,k1,w,o] e^{+2pi i k1 h/128} ) / 16384
// ---------------------------------------------------------------------------
__global__ __launch_bounds__(256)
void k_idfth(const float* __restrict__ Tre, const float* __restrict__ Tim,
             float* __restrict__ out) {
  const int t = blockIdx.x * 256 + threadIdx.x;        // [b][h][w][o]
  const int o = t & 63, w = (t >> 6) & 127, h = (t >> 13) & 127, b = t >> 20;
  const int tbase = (b * MODES * HW + w) * 64 + o;     // + k1 * HW*64
  float acc = 0.f;
#pragma unroll
  for (int k1 = 0; k1 < MODES; ++k1) {
    float s, cc;
    __sincosf((float)((k1 * h) & 127) * TWO_PI_OVER_128, &s, &cc);
    const int idx = tbase + k1 * (HW * 64);
    acc = fmaf(Tre[idx], cc, fmaf(-Tim[idx], s, acc));
  }
  out[t] = acc * (1.0f / 16384.0f);
}

// ---------------------------------------------------------------------------
extern "C" void kernel_launch(void* const* d_in, const int* in_sizes, int n_in,
                              void* d_out, int out_size, void* d_ws, size_t ws_size,
                              hipStream_t stream) {
  (void)in_sizes; (void)n_in; (void)out_size; (void)ws_size;
  const float* x  = (const float*)d_in[0];
  const float* wr = (const float*)d_in[1];
  const float* wi = (const float*)d_in[2];
  float* out = (float*)d_out;

  char* ws = (char*)d_ws;                    // ~13 MB used
  unsigned short* Wr16 = (unsigned short*)(ws + 0);                    // 2 MB
  unsigned short* Wi16 = (unsigned short*)(ws + (2u  << 20));          // 2 MB
  float* Yre           = (float*)(ws + (4u  << 20));                   // 2 MB
  float* Yim           = (float*)(ws + (6u  << 20));                   // 2 MB
  unsigned short* Xr16 = (unsigned short*)(ws + (8u  << 20));          // 128 KB
  unsigned short* Xi16 = (unsigned short*)(ws + (8u  << 20) + (128u << 10));
  float* Zre           = (float*)(ws + (8u  << 20) + (512u << 10));    // 256 KB
  float* Zim           = (float*)(ws + (8u  << 20) + (768u << 10));    // 256 KB
  float* Tre           = (float*)(ws + (9u  << 20));                   // 2 MB
  float* Tim           = (float*)(ws + (11u << 20));                   // 2 MB

  k_sumw <<<1024,  256, 0, stream>>>(wr, wi, Wr16, Wi16);     // 33.5 MB stream (the runtime)
  k_dftw <<<2048,  256, 0, stream>>>(x, Yre, Yim);
  k_dfth <<<256,   256, 0, stream>>>(Yre, Yim, Xr16, Xi16);
  k_gemm <<<32,    256, 0, stream>>>(Xr16, Xi16, Wr16, Wi16, Zre, Zim);
  k_idftm<<<2048,  256, 0, stream>>>(Zre, Zim, Tre, Tim);
  k_idfth<<<16384, 256, 0, stream>>>(Tre, Tim, out);
}
